// Krop_81544249082422
// MI455X (gfx1250) — matrix operand order repaired and checked by reading, not compiled, not run on hardware
//
#include <hip/hip_runtime.h>
#include <hip/hip_bf16.h>
#include <math.h>

// ---------------------------------------------------------------------------
// MI455X (gfx1250) transformer forward: bf16 WMMA GEMMs + wave32 flash attn.
// Compute-bound (~480 GFLOP dense); weights (192MB fp32 -> 96MB bf16) are
// L2-resident, so everything targets v_wmma_f32_16x16x32_bf16 throughput.
// GEMM uses double-buffered GLOBAL_LOAD_ASYNC_TO_LDS_B128 (ASYNCcnt) so tile
// fills bypass VGPRs and overlap WMMA compute; one barrier per K-step.
// ---------------------------------------------------------------------------

typedef __bf16 bf16;
typedef __attribute__((ext_vector_type(16))) __bf16 v16bf;
typedef __attribute__((ext_vector_type(8)))  __bf16 v8bf;
typedef __attribute__((ext_vector_type(8)))  float  v8f;

union V16 { v16bf v; v8bf h[2]; };

#define WMMA_BF16(a, b, c) \
    __builtin_amdgcn_wmma_f32_16x16x32_bf16(false, (a), false, (b), (short)0, (c), false, false)

// Async global->LDS copy (CDNA5 VFLAT async op, tracked by ASYNCcnt).
__device__ __forceinline__ void async_ld_b128(void* lds, const void* g)
{
#if defined(__HIP_DEVICE_COMPILE__)
    unsigned l = (unsigned)(unsigned long long)(size_t)lds;   // LDS byte offset
    unsigned long long ga = (unsigned long long)(size_t)g;    // 64-bit global addr
    asm volatile("global_load_async_to_lds_b128 %0, %1, off"
                 :: "v"(l), "v"(ga) : "memory");
#endif
}

__device__ __forceinline__ void wait_async0()
{
#if defined(__HIP_DEVICE_COMPILE__)
    asm volatile("s_wait_asynccnt 0x0" ::: "memory");
#endif
}

// ---------------------------------------------------------------------------
// Generic GEMM: C[M,N] = A[M,K](bf16, row-major) x Bt[N,K](bf16, pre-transposed)
// Block = 256 threads = 8 waves; block tile 128x128; K-step 32.
// Wave tile 32x64 = 2x4 accumulators of 16x16. LDS double-buffered via async.
// ---------------------------------------------------------------------------
#define GT_M 128
#define GT_N 128
#define GT_K 32
#define LDK  40   // padded LDS stride in bf16 (80B rows: 16B-aligned, conflict-shifted)

__global__ __launch_bounds__(256) void gemm_bf16_wmma(
    const bf16* __restrict__ A, const bf16* __restrict__ Bt,
    float* __restrict__ Cf, bf16* __restrict__ Cb,
    int M, int N, int K)
{
    __shared__ __align__(16) bf16 As[2][GT_M * LDK];
    __shared__ __align__(16) bf16 Bs[2][GT_N * LDK];

    const int t    = threadIdx.x;
    const int lane = t & 31;
    const int half = lane >> 4;
    const int lm   = lane & 15;
    const int wave = t >> 5;
    const int wm   = (wave & 3) * 32;   // 4 waves along M
    const int wn   = (wave >> 2) * 64;  // 2 waves along N
    const int bM   = blockIdx.y * GT_M;
    const int bN   = blockIdx.x * GT_N;

    // Two 8-bf16 (128-bit) chunks per thread per tile per matrix.
    const int c1  = t + 256;
    const int r0  = t >> 2,  kc0 = (t & 3) << 3;
    const int r1  = c1 >> 2, kc1 = (c1 & 3) << 3;

    auto stage = [&](int buf, int k0) {
        async_ld_b128(&As[buf][r0 * LDK + kc0], &A[(size_t)(bM + r0) * K + k0 + kc0]);
        async_ld_b128(&Bs[buf][r0 * LDK + kc0], &Bt[(size_t)(bN + r0) * K + k0 + kc0]);
        async_ld_b128(&As[buf][r1 * LDK + kc1], &A[(size_t)(bM + r1) * K + k0 + kc1]);
        async_ld_b128(&Bs[buf][r1 * LDK + kc1], &Bt[(size_t)(bN + r1) * K + k0 + kc1]);
    };

    v8f acc[2][4];
#pragma unroll
    for (int mi = 0; mi < 2; ++mi)
#pragma unroll
        for (int nj = 0; nj < 4; ++nj)
#pragma unroll
            for (int e = 0; e < 8; ++e) acc[mi][nj][e] = 0.f;

    const int nstep = K / GT_K;
    stage(0, 0);

    for (int s = 0; s < nstep; ++s) {
        // Own async fills of buffer (s&1) done, then barrier => everyone's done.
        wait_async0();
        __syncthreads();
        if (s + 1 < nstep) stage((s + 1) & 1, (s + 1) * GT_K);
        if (s + 2 < nstep) {
            __builtin_prefetch(&A[(size_t)(bM + r0) * K + (s + 2) * GT_K + kc0], 0, 2);
            __builtin_prefetch(&Bt[(size_t)(bN + r0) * K + (s + 2) * GT_K + kc0], 0, 2);
        }

        const bf16* as = As[s & 1];
        const bf16* bs = Bs[s & 1];

        // A fragment (16x32): lane row = lm; K groups at 8*half and 16+8*half.
        V16 a[2], b[4];
#pragma unroll
        for (int mi = 0; mi < 2; ++mi) {
            const bf16* ap = &as[(wm + mi * 16 + lm) * LDK];
            a[mi].h[0] = *(const v8bf*)&ap[8 * half];
            a[mi].h[1] = *(const v8bf*)&ap[16 + 8 * half];
        }
        // B fragment (32x16): lane col = lm; K groups at 8*half and 16+8*half.
#pragma unroll
        for (int nj = 0; nj < 4; ++nj) {
            const bf16* bp = &bs[(wn + nj * 16 + lm) * LDK];
            b[nj].h[0] = *(const v8bf*)&bp[8 * half];
            b[nj].h[1] = *(const v8bf*)&bp[16 + 8 * half];
        }
#pragma unroll
        for (int mi = 0; mi < 2; ++mi)
#pragma unroll
            for (int nj = 0; nj < 4; ++nj)
                acc[mi][nj] = WMMA_BF16(a[mi].v, b[nj].v, acc[mi][nj]);
    }

    // Epilogue: C layout VGPR v -> row = v + 8*half, col = lm.
    if (Cf) {
#pragma unroll
        for (int mi = 0; mi < 2; ++mi)
#pragma unroll
            for (int nj = 0; nj < 4; ++nj)
#pragma unroll
                for (int v = 0; v < 8; ++v) {
                    int row = bM + wm + mi * 16 + v + 8 * half;
                    int col = bN + wn + nj * 16 + lm;
                    Cf[(size_t)row * N + col] = acc[mi][nj][v];
                }
    } else {
#pragma unroll
        for (int mi = 0; mi < 2; ++mi)
#pragma unroll
            for (int nj = 0; nj < 4; ++nj)
#pragma unroll
                for (int v = 0; v < 8; ++v) {
                    int row = bM + wm + mi * 16 + v + 8 * half;
                    int col = bN + wn + nj * 16 + lm;
                    Cb[(size_t)row * N + col] = (bf16)acc[mi][nj][v];
                }
    }
}

// ---------------------------------------------------------------------------
// Weight transpose + fp32->bf16: src[K,N] f32 -> dst[N,K] bf16.
// ---------------------------------------------------------------------------
__global__ void transpose_f32_bf16(const float* __restrict__ src,
                                   bf16* __restrict__ dst, int K, int N)
{
    __shared__ float tile[32][33];
    int tx = threadIdx.x;
    int x = blockIdx.x * 32 + tx;      // N index
    int y0 = blockIdx.y * 32;          // K base
    for (int j = threadIdx.y; j < 32; j += 8) {
        int y = y0 + j;
        tile[j][tx] = (y < K && x < N) ? src[(size_t)y * N + x] : 0.f;
    }
    __syncthreads();
    int xo = y0 + tx;                  // K index in output
    int yo0 = blockIdx.x * 32;         // N base in output
    for (int j = threadIdx.y; j < 32; j += 8) {
        int yo = yo0 + j;
        if (yo < N && xo < K) dst[(size_t)yo * K + xo] = (bf16)tile[tx][j];
    }
}

// ---------------------------------------------------------------------------
// RMSNorm over D=1024, bf16 output (feeds GEMM A operand).
// ---------------------------------------------------------------------------
__global__ __launch_bounds__(256) void rmsnorm_bf16_k(
    const float* __restrict__ x, const float* __restrict__ w,
    bf16* __restrict__ out, int D)
{
    const int row = blockIdx.x;
    const float* xr = x + (size_t)row * D;
    const int t = threadIdx.x;
    float v[4], s = 0.f;
#pragma unroll
    for (int i = 0; i < 4; ++i) { v[i] = xr[t + 256 * i]; s += v[i] * v[i]; }
#pragma unroll
    for (int o = 16; o > 0; o >>= 1) s += __shfl_xor(s, o, 32);
    __shared__ float wsum[8];
    if ((t & 31) == 0) wsum[t >> 5] = s;
    __syncthreads();
    float tot = 0.f;
#pragma unroll
    for (int i = 0; i < 8; ++i) tot += wsum[i];
    float r = rsqrtf(tot / (float)D + 1e-6f);
#pragma unroll
    for (int i = 0; i < 4; ++i) {
        int c = t + 256 * i;
        out[(size_t)row * D + c] = (bf16)(v[i] * r * w[c]);
    }
}

// Final norm -> fp32 output.
__global__ __launch_bounds__(256) void rmsnorm_f32_k(
    const float* __restrict__ x, const float* __restrict__ w,
    float* __restrict__ out, int D)
{
    const int row = blockIdx.x;
    const float* xr = x + (size_t)row * D;
    const int t = threadIdx.x;
    float v[4], s = 0.f;
#pragma unroll
    for (int i = 0; i < 4; ++i) { v[i] = xr[t + 256 * i]; s += v[i] * v[i]; }
#pragma unroll
    for (int o = 16; o > 0; o >>= 1) s += __shfl_xor(s, o, 32);
    __shared__ float wsum[8];
    if ((t & 31) == 0) wsum[t >> 5] = s;
    __syncthreads();
    float tot = 0.f;
#pragma unroll
    for (int i = 0; i < 8; ++i) tot += wsum[i];
    float r = rsqrtf(tot / (float)D + 1e-6f);
#pragma unroll
    for (int i = 0; i < 4; ++i) {
        int c = t + 256 * i;
        out[(size_t)row * D + c] = v[i] * r * w[c];
    }
}

// ---------------------------------------------------------------------------
// Per-head RMSNorm (over HD=64) + RoPE; f32 in, bf16 out (attention operands).
// Block = NH*16 threads; one token per block; 16 threads per head.
// ---------------------------------------------------------------------------
__global__ void qknorm_rope_k(const float* __restrict__ x,
                              const float* __restrict__ w,
                              bf16* __restrict__ out, int NH, int S)
{
    const int token = blockIdx.x;
    const int t = threadIdx.x;
    const int head = t >> 4, l16 = t & 15;
    const float* xr = x + (size_t)token * NH * 64 + head * 64;
    const int d0 = l16 * 4;
    float v[4], s = 0.f;
#pragma unroll
    for (int i = 0; i < 4; ++i) { v[i] = xr[d0 + i]; s += v[i] * v[i]; }
#pragma unroll
    for (int o = 8; o > 0; o >>= 1) s += __shfl_xor(s, o, 16);
    float r = rsqrtf(s / 64.f + 1e-6f);
    const int pos = token & (S - 1);
    bf16* orow = out + (size_t)token * NH * 64 + head * 64;
#pragma unroll
    for (int i = 0; i < 4; ++i) {
        int d = d0 + i;
        int dp = (d < 32) ? d + 32 : d - 32;
        float val = v[i] * r * w[d];
        float pv = xr[dp] * r * w[dp];
        float rot = (d < 32) ? -pv : pv;
        // inv_freq = theta^(-(d%32)/32), theta = 1e6, ln(1e6)=13.8155105...
        float inv = __expf((float)(d & 31) * (-13.815510557964274f / 32.f));
        float ang = (float)pos * inv;
        orow[d] = (bf16)(val * __cosf(ang) + rot * __sinf(ang));
    }
}

// ---------------------------------------------------------------------------
// Flash attention, one wave32 per (b, h, 16-query tile). WMMA for QK^T and PV.
// sliding != 0 -> band |q-k| <= 12 only.
// ---------------------------------------------------------------------------
__global__ __launch_bounds__(32) void attn_wmma_k(
    const bf16* __restrict__ qb, const bf16* __restrict__ kb,
    const bf16* __restrict__ vb, const int* __restrict__ amask,
    bf16* __restrict__ ctx, int sliding)
{
    const int S = 1024, H = 16, HKc = 8, HD = 64;
    const int qt = blockIdx.x, h = blockIdx.y, b = blockIdx.z;
    const int kvh = h >> 1;         // H/HK = 2
    const int q0 = qt * 16;
    const int lane = threadIdx.x;
    const int half = lane >> 4, lm = lane & 15;

    __shared__ __align__(16) bf16 Qs[16 * 64];
    __shared__ __align__(16) bf16 Ks[32 * 64];
    __shared__ __align__(16) bf16 Vt[64 * 32];   // V transposed: [feat][key]
    __shared__ __align__(16) bf16 Ps[16 * 32];
    __shared__ float mk[32];

    // Load 16x64 Q tile.
#pragma unroll
    for (int i = 0; i < 4; ++i) {
        int c = lane + 32 * i;
        int r = c >> 3, kc = (c & 7) * 8;
        *(v8bf*)&Qs[r * 64 + kc] =
            *(const v8bf*)&qb[(size_t)(b * S + q0 + r) * (H * HD) + h * HD + kc];
    }
    __syncthreads();
    V16 aq[2];
#pragma unroll
    for (int c = 0; c < 2; ++c) {
        const bf16* qp = &Qs[lm * 64 + c * 32];
        aq[c].h[0] = *(const v8bf*)&qp[8 * half];
        aq[c].h[1] = *(const v8bf*)&qp[16 + 8 * half];
    }

    float mrow[8], lrow[8];
    v8f o[4];
#pragma unroll
    for (int v = 0; v < 8; ++v) { mrow[v] = -3.0e38f; lrow[v] = 0.f; }
#pragma unroll
    for (int nj = 0; nj < 4; ++nj)
#pragma unroll
        for (int v = 0; v < 8; ++v) o[nj][v] = 0.f;

    int kt0 = 0, kt1 = S;
    if (sliding) {
        int a = q0 - 12; kt0 = (a > 0 ? a : 0) & ~31;
        int e = q0 + 16 + 12; kt1 = e < S ? e : S;
    }

    for (int kt = kt0; kt < kt1; kt += 32) {
        __syncthreads();
        // 32x64 K tile.
#pragma unroll
        for (int i = 0; i < 8; ++i) {
            int c = lane + 32 * i;
            int r = c >> 3, kc = (c & 7) * 8;
            *(v8bf*)&Ks[r * 64 + kc] =
                *(const v8bf*)&kb[(size_t)(b * S + kt + r) * (HKc * HD) + kvh * HD + kc];
        }
        // 32x64 V tile, stored transposed; lane owns one key row.
        {
            const bf16* vr = &vb[(size_t)(b * S + kt + lane) * (HKc * HD) + kvh * HD];
#pragma unroll
            for (int f = 0; f < 64; ++f) Vt[f * 32 + lane] = vr[f];
        }
        mk[lane] = (float)amask[(size_t)b * S + kt + lane];
        __syncthreads();

        // Scores: two 16x16 tiles, K(feature)=64 -> 2 WMMAs each.
        v8f sc[2];
#pragma unroll
        for (int j = 0; j < 2; ++j)
#pragma unroll
            for (int v = 0; v < 8; ++v) sc[j][v] = 0.f;
#pragma unroll
        for (int j = 0; j < 2; ++j)
#pragma unroll
            for (int c = 0; c < 2; ++c) {
                V16 bk;
                const bf16* kp = &Ks[(j * 16 + lm) * 64 + c * 32];
                bk.h[0] = *(const v8bf*)&kp[8 * half];
                bk.h[1] = *(const v8bf*)&kp[16 + 8 * half];
                sc[j] = WMMA_BF16(aq[c].v, bk.v, sc[j]);
            }

        // Online softmax per row (row = q0 + v + 8*half; 16 lanes share a row).
#pragma unroll
        for (int v = 0; v < 8; ++v) {
            int row = q0 + v + 8 * half;
            float best = -3.0e38f;
            float pj[2];
#pragma unroll
            for (int j = 0; j < 2; ++j) {
                int key = kt + j * 16 + lm;
                float sv = sc[j][v] * 0.125f;   // 1/sqrt(64)
                bool ok = (mk[j * 16 + lm] != 0.f);
                if (sliding) { int dlt = key - row; if (dlt < 0) dlt = -dlt; ok = ok && (dlt <= 12); }
                if (!ok) sv = -3.0e38f;
                sc[j][v] = sv;
                best = best > sv ? best : sv;
            }
#pragma unroll
            for (int off = 1; off < 16; off <<= 1) {
                float o2 = __shfl_xor(best, off, 32);
                best = best > o2 ? best : o2;
            }
            float mnew = mrow[v] > best ? mrow[v] : best;
            float corr = __expf(mrow[v] - mnew);
            mrow[v] = mnew;
            float rs = 0.f;
#pragma unroll
            for (int j = 0; j < 2; ++j) {
                float p = __expf(sc[j][v] - mnew);
                pj[j] = p; rs += p;
            }
#pragma unroll
            for (int off = 1; off < 16; off <<= 1) rs += __shfl_xor(rs, off, 32);
            lrow[v] = lrow[v] * corr + rs;
#pragma unroll
            for (int nj = 0; nj < 4; ++nj) o[nj][v] *= corr;
            Ps[(v + 8 * half) * 32 + lm] = (bf16)pj[0];
            Ps[(v + 8 * half) * 32 + 16 + lm] = (bf16)pj[1];
        }
        __syncthreads();

        // PV: P(16x32, A) x V(32x64, B) -> 4 output tiles.
        V16 ap;
        {
            const bf16* pp = &Ps[lm * 32];
            ap.h[0] = *(const v8bf*)&pp[8 * half];
            ap.h[1] = *(const v8bf*)&pp[16 + 8 * half];
        }
#pragma unroll
        for (int nj = 0; nj < 4; ++nj) {
            V16 bv;
            const bf16* vp = &Vt[(nj * 16 + lm) * 32];
            bv.h[0] = *(const v8bf*)&vp[8 * half];
            bv.h[1] = *(const v8bf*)&vp[16 + 8 * half];
            o[nj] = WMMA_BF16(ap.v, bv.v, o[nj]);
        }
    }

    // Normalize and write ctx (bf16, feeds the wo GEMM).
#pragma unroll
    for (int nj = 0; nj < 4; ++nj)
#pragma unroll
        for (int v = 0; v < 8; ++v) {
            int row = q0 + v + 8 * half;
            int col = h * HD + nj * 16 + lm;
            ctx[(size_t)(b * S + row) * (H * HD) + col] = (bf16)(o[nj][v] / lrow[v]);
        }
}

// ---------------------------------------------------------------------------
// Elementwise helpers.
// ---------------------------------------------------------------------------
__global__ void add_inplace_k(float* __restrict__ h, const float* __restrict__ t, int n)
{
    int i = blockIdx.x * 256 + threadIdx.x;
    if (i < n) h[i] += t[i];
}

__global__ void silu_mul_k(const float* __restrict__ g, const float* __restrict__ u,
                           bf16* __restrict__ o, int n)
{
    int i = blockIdx.x * 256 + threadIdx.x;
    if (i < n) {
        float x = g[i];
        o[i] = (bf16)((x / (1.f + __expf(-x))) * u[i]);
    }
}

// ---------------------------------------------------------------------------
// Orchestration.
// ---------------------------------------------------------------------------
extern "C" void kernel_launch(void* const* d_in, const int* in_sizes, int n_in,
                              void* d_out, int out_size, void* d_ws, size_t ws_size,
                              hipStream_t stream)
{
    const int Lc = 4, D = 1024, H = 16, HKc = 8, HD = 64, FF = 3072, B = 4, S = 1024;
    const int T = B * S;  // 4096 tokens
    const int DQ = H * HD;   // 1024
    const int DK = HKc * HD; // 512

    const float* emb   = (const float*)d_in[0];
    const float* wq    = (const float*)d_in[1];
    const float* wk    = (const float*)d_in[2];
    const float* wv    = (const float*)d_in[3];
    const float* wo    = (const float*)d_in[4];
    const float* qnw   = (const float*)d_in[5];
    const float* knw   = (const float*)d_in[6];
    const float* ln1   = (const float*)d_in[7];
    const float* ln2   = (const float*)d_in[8];
    const float* wg    = (const float*)d_in[9];
    const float* wu    = (const float*)d_in[10];
    const float* wd    = (const float*)d_in[11];
    const float* normw = (const float*)d_in[12];
    const int*   amask = (const int*)d_in[13];
    float* outp = (float*)d_out;

    char* ws = (char*)d_ws;
    size_t off = 0;
    auto alloc = [&](size_t bytes) -> char* {
        char* p = ws + off;
        off += (bytes + 255) & ~(size_t)255;
        return p;
    };

    bf16* wqT = (bf16*)alloc((size_t)Lc * DQ * D * 2);
    bf16* wkT = (bf16*)alloc((size_t)Lc * DK * D * 2);
    bf16* wvT = (bf16*)alloc((size_t)Lc * DK * D * 2);
    bf16* woT = (bf16*)alloc((size_t)Lc * D * DQ * 2);
    bf16* wgT = (bf16*)alloc((size_t)Lc * FF * D * 2);
    bf16* wuT = (bf16*)alloc((size_t)Lc * FF * D * 2);
    bf16* wdT = (bf16*)alloc((size_t)Lc * D * FF * 2);
    float* hbuf  = (float*)alloc((size_t)T * D * 4);
    bf16*  nbf   = (bf16*)alloc((size_t)T * D * 2);
    float* qf    = (float*)alloc((size_t)T * DQ * 4);
    float* kf    = (float*)alloc((size_t)T * DK * 4);
    bf16*  qbf   = (bf16*)alloc((size_t)T * DQ * 2);
    bf16*  kbf   = (bf16*)alloc((size_t)T * DK * 2);
    bf16*  vbf   = (bf16*)alloc((size_t)T * DK * 2);
    bf16*  ctxb  = (bf16*)alloc((size_t)T * DQ * 2);
    float* tf    = (float*)alloc((size_t)T * D * 4);
    float* gatef = (float*)alloc((size_t)T * FF * 4);
    float* upf   = (float*)alloc((size_t)T * FF * 4);
    bf16*  actb  = (bf16*)alloc((size_t)T * FF * 2);

    dim3 tb(32, 8);
    // Transpose + bf16-convert all weights once (L2-resident thereafter).
    for (int l = 0; l < Lc; ++l) {
        transpose_f32_bf16<<<dim3(DQ / 32, D / 32), tb, 0, stream>>>(
            wq + (size_t)l * D * DQ, wqT + (size_t)l * DQ * D, D, DQ);
        transpose_f32_bf16<<<dim3(DK / 32, D / 32), tb, 0, stream>>>(
            wk + (size_t)l * D * DK, wkT + (size_t)l * DK * D, D, DK);
        transpose_f32_bf16<<<dim3(DK / 32, D / 32), tb, 0, stream>>>(
            wv + (size_t)l * D * DK, wvT + (size_t)l * DK * D, D, DK);
        transpose_f32_bf16<<<dim3(D / 32, DQ / 32), tb, 0, stream>>>(
            wo + (size_t)l * DQ * D, woT + (size_t)l * D * DQ, DQ, D);
        transpose_f32_bf16<<<dim3(FF / 32, D / 32), tb, 0, stream>>>(
            wg + (size_t)l * D * FF, wgT + (size_t)l * FF * D, D, FF);
        transpose_f32_bf16<<<dim3(FF / 32, D / 32), tb, 0, stream>>>(
            wu + (size_t)l * D * FF, wuT + (size_t)l * FF * D, D, FF);
        transpose_f32_bf16<<<dim3(D / 32, FF / 32), tb, 0, stream>>>(
            wd + (size_t)l * FF * D, wdT + (size_t)l * D * FF, FF, D);
    }

    // h = inputs_embeds
    hipMemcpyAsync(hbuf, emb, (size_t)T * D * sizeof(float),
                   hipMemcpyDeviceToDevice, stream);

    const int nTD = T * D, nTF = T * FF;
    for (int l = 0; l < Lc; ++l) {
        // --- attention block ---
        rmsnorm_bf16_k<<<T, 256, 0, stream>>>(hbuf, ln1 + (size_t)l * D, nbf, D);
        gemm_bf16_wmma<<<dim3(DQ / GT_N, T / GT_M), 256, 0, stream>>>(
            nbf, wqT + (size_t)l * DQ * D, qf, nullptr, T, DQ, D);
        gemm_bf16_wmma<<<dim3(DK / GT_N, T / GT_M), 256, 0, stream>>>(
            nbf, wkT + (size_t)l * DK * D, kf, nullptr, T, DK, D);
        gemm_bf16_wmma<<<dim3(DK / GT_N, T / GT_M), 256, 0, stream>>>(
            nbf, wvT + (size_t)l * DK * D, nullptr, vbf, T, DK, D);
        qknorm_rope_k<<<T, H * 16, 0, stream>>>(qf, qnw + (size_t)l * HD, qbf, H, S);
        qknorm_rope_k<<<T, HKc * 16, 0, stream>>>(kf, knw + (size_t)l * HD, kbf, HKc, S);
        attn_wmma_k<<<dim3(S / 16, H, B), 32, 0, stream>>>(
            qbf, kbf, vbf, amask, ctxb, l & 1);
        gemm_bf16_wmma<<<dim3(D / GT_N, T / GT_M), 256, 0, stream>>>(
            ctxb, woT + (size_t)l * D * DQ, tf, nullptr, T, D, DQ);
        add_inplace_k<<<(nTD + 255) / 256, 256, 0, stream>>>(hbuf, tf, nTD);

        // --- MLP block ---
        rmsnorm_bf16_k<<<T, 256, 0, stream>>>(hbuf, ln2 + (size_t)l * D, nbf, D);
        gemm_bf16_wmma<<<dim3(FF / GT_N, T / GT_M), 256, 0, stream>>>(
            nbf, wgT + (size_t)l * FF * D, gatef, nullptr, T, FF, D);
        gemm_bf16_wmma<<<dim3(FF / GT_N, T / GT_M), 256, 0, stream>>>(
            nbf, wuT + (size_t)l * FF * D, upf, nullptr, T, FF, D);
        silu_mul_k<<<(nTF + 255) / 256, 256, 0, stream>>>(gatef, upf, actb, nTF);
        gemm_bf16_wmma<<<dim3(D / GT_N, T / GT_M), 256, 0, stream>>>(
            actb, wdT + (size_t)l * D * FF, tf, nullptr, T, D, FF);
        add_inplace_k<<<(nTD + 255) / 256, 256, 0, stream>>>(hbuf, tf, nTD);
    }

    rmsnorm_f32_k<<<T, 256, 0, stream>>>(hbuf, normw, outp, D);
}
